// JPEGEncoder_481036337612
// MI455X (gfx1250) — matrix operand order repaired and checked by reading, not compiled, not run on hardware
//
#include <hip/hip_runtime.h>
#include <hip/hip_bf16.h>

typedef __attribute__((ext_vector_type(16))) _Float16 v16h;
typedef __attribute__((ext_vector_type(8)))  float    v8f;

// Reciprocals of the quant tables — each entry constant-folded by the compiler,
// so quantization is one v_mul instead of an IEEE divide sequence.
__constant__ float c_RQY[64] = {
 1.f/16,1.f/11,1.f/10,1.f/16,1.f/24,1.f/40,1.f/51,1.f/61,
 1.f/12,1.f/12,1.f/14,1.f/19,1.f/26,1.f/58,1.f/60,1.f/55,
 1.f/14,1.f/13,1.f/16,1.f/24,1.f/40,1.f/57,1.f/69,1.f/56,
 1.f/14,1.f/17,1.f/22,1.f/29,1.f/51,1.f/87,1.f/80,1.f/62,
 1.f/18,1.f/22,1.f/37,1.f/56,1.f/68,1.f/109,1.f/103,1.f/77,
 1.f/24,1.f/35,1.f/55,1.f/64,1.f/81,1.f/104,1.f/113,1.f/92,
 1.f/49,1.f/64,1.f/78,1.f/87,1.f/103,1.f/121,1.f/120,1.f/101,
 1.f/72,1.f/92,1.f/95,1.f/98,1.f/112,1.f/100,1.f/103,1.f/99};
__constant__ float c_RQC[64] = {
 1.f/17,1.f/18,1.f/24,1.f/47,1.f/99,1.f/99,1.f/99,1.f/99,
 1.f/18,1.f/21,1.f/26,1.f/66,1.f/99,1.f/99,1.f/99,1.f/99,
 1.f/24,1.f/26,1.f/56,1.f/99,1.f/99,1.f/99,1.f/99,1.f/99,
 1.f/47,1.f/66,1.f/99,1.f/99,1.f/99,1.f/99,1.f/99,1.f/99,
 1.f/99,1.f/99,1.f/99,1.f/99,1.f/99,1.f/99,1.f/99,1.f/99,
 1.f/99,1.f/99,1.f/99,1.f/99,1.f/99,1.f/99,1.f/99,1.f/99,
 1.f/99,1.f/99,1.f/99,1.f/99,1.f/99,1.f/99,1.f/99,1.f/99,
 1.f/99,1.f/99,1.f/99,1.f/99,1.f/99,1.f/99,1.f/99,1.f/99};

// DCT-II basis: M[x][u] = cos((2x+1)*u*pi/16)
__device__ __forceinline__ float dctM(int x, int u) {
  return __cosf((float)((2 * x + 1) * u) * (3.14159265358979323846f / 16.0f));
}

// One 16x16 tile = 2x2 blocks of 8x8.  R = diag(Mt,Mt) * X * diag(M,M).
// aMt has lanes 16..31 == 0, so every K>=16 product vanishes regardless of
// the (finite) garbage in B's upper-lane K-half — no B-side masking needed.
__device__ __forceinline__ v8f dct16(v16h bX, v16h aMt, v16h bM,
                                     float* sT, int lane) {
  const int h  = lane >> 4;
  const int lc = lane & 15;
  v8f zero = {};
  // T = diag(Mt,Mt) * X   (f32 accumulate)
  v8f T = __builtin_amdgcn_wmma_f32_16x16x32_f16(
      false, aMt, false, bX, (short)0, zero, false, false);
  __syncthreads();                 // WAR vs previous tile's scratch reads
#pragma unroll
  for (int i = 0; i < 8; ++i)      // D-layout: lane holds T[i+8h][lc]
    sT[(i + 8 * h) * 17 + lc] = T[i];
  __syncthreads();
  v16h aT;                         // A-layout: lanes 0-15 need T[lc][i]; lanes 16-31 zero (K>=16)
#pragma unroll
  for (int i = 0; i < 16; ++i)
    aT[i] = h ? (_Float16)0.0f : (_Float16)sT[lc * 17 + i];
  // R = T * diag(M,M)
  return __builtin_amdgcn_wmma_f32_16x16x32_f16(
      false, aT, false, bM, (short)0, zero, false, false);
}

__global__ __launch_bounds__(32)
void jpeg_enc_kernel(const float* __restrict__ x, float* __restrict__ out) {
  __shared__ float sY [32 * 33];   // Y tile (value-128), padded rows
  __shared__ float sCb[32 * 17];   // horizontal 2-pair sums of Cb (no shift)
  __shared__ float sCr[32 * 17];
  __shared__ float sT [16 * 17];   // f32 intermediate scratch

  const int lane = threadIdx.x;
  const int b  = blockIdx.x >> 8;          // image
  const int t  = blockIdx.x & 255;         // 32x32 tile within 512x512
  const int wy = t >> 4, wx = t & 15;
  const int h  = lane >> 4;                // block-row within 16x16 tile
  const int lc = lane & 15;                // tile column
  const int bc = (lane >> 3) & 1;          // block-col within 16x16 tile
  const int vq = lane & 7;                 // v index within 8x8 block

  // ---- per-lane constant WMMA operands --------------------------------
  v16h aMt, bM;
#pragma unroll
  for (int i = 0; i < 16; ++i) {
    // A: Mt2[r][c], r = lc, c = i (lanes 16-31 zero-pad K)
    _Float16 av = (_Float16)0.0f;
    if (h == 0) {
      int r = lc, c = i;
      float mv = ((r >> 3) == (c >> 3)) ? dctM(c & 7, r & 7) : 0.0f; // Mt[r][c]=M[c][r]
      av = (_Float16)mv;
    }
    aMt[i] = av;
    // B: M2[i][j], j = lc; upper-lane K-half left unmasked (multiplied by 0).
    float bv = ((i >> 3) == (lc >> 3)) ? dctM(i & 7, lc & 7) : 0.0f;
    bM[i] = (_Float16)bv;
  }
  // quant coefficients: SCALE[u][v] * (1/Q[u][v])   (FACTOR = 1)
  float cY[8], cC[8];
  {
    const float inv_sqrt2 = 0.70710678118654752440f;
    float av_v = (vq == 0) ? inv_sqrt2 : 1.0f;
#pragma unroll
    for (int i = 0; i < 8; ++i) {
      float au = (i == 0) ? inv_sqrt2 : 1.0f;
      float s = au * av_v * 0.25f;
      cY[i] = s * c_RQY[i * 8 + vq];
      cC[i] = s * c_RQC[i * 8 + vq];
    }
  }

  // ---- load 32x32 RGB tile: color convert + chroma pair-sums ----------
  const size_t plane = (size_t)512 * 512;
  const float* pR = x + (size_t)b * 3 * plane + (size_t)(wy * 32 + lane) * 512 + wx * 32;
  const float* pG = pR + plane;
  const float* pB = pG + plane;
#pragma unroll
  for (int c = 0; c < 8; ++c) {
    float4 r4 = reinterpret_cast<const float4*>(pR)[c];
    float4 g4 = reinterpret_cast<const float4*>(pG)[c];
    float4 b4 = reinterpret_cast<const float4*>(pB)[c];
    float rr[4] = {r4.x, r4.y, r4.z, r4.w};
    float gg[4] = {g4.x, g4.y, g4.z, g4.w};
    float bb[4] = {b4.x, b4.y, b4.z, b4.w};
    float cb[4], cr[4];
#pragma unroll
    for (int k = 0; k < 4; ++k) {
      float yv = 0.299f * rr[k] + 0.587f * gg[k] + 0.114f * bb[k];
      cb[k] = -0.168736f * rr[k] - 0.331264f * gg[k] + 0.5f * bb[k];
      cr[k] = 0.5f * rr[k] - 0.418688f * gg[k] - 0.081312f * bb[k];
      sY[lane * 33 + 4 * c + k] = yv - 128.0f;   // (+0 shift) - 128
    }
    // chroma: (+128 shift) - 128 cancels; horizontal pair-sums
    sCb[lane * 17 + 2 * c + 0] = cb[0] + cb[1];
    sCb[lane * 17 + 2 * c + 1] = cb[2] + cb[3];
    sCr[lane * 17 + 2 * c + 0] = cr[0] + cr[1];
    sCr[lane * 17 + 2 * c + 1] = cr[2] + cr[3];
  }
  __syncthreads();

  float* yq  = out;
  float* cbq = out + (size_t)32 * 4096 * 64;
  float* crq = cbq + (size_t)32 * 1024 * 64;

  // ---- 4 luma tiles (each = 2x2 blocks of 8x8) ------------------------
#pragma unroll
  for (int sy = 0; sy < 2; ++sy)
#pragma unroll
    for (int sx = 0; sx < 2; ++sx) {
      v16h bX;
#pragma unroll
      for (int i = 0; i < 16; ++i) {
        float v = sY[(sy * 16 + i) * 33 + sx * 16 + lc];
        bX[i] = (_Float16)v;       // upper lanes: harmless (x0 in A)
      }
      v8f R = dct16(bX, aMt, bM, sT, lane);
      int by = wy * 4 + sy * 2 + h;
      int bx = wx * 4 + sx * 2 + bc;
      float* op = yq + (((size_t)b * 4096) + (size_t)by * 64 + bx) * 64 + vq;
#pragma unroll
      for (int i = 0; i < 8; ++i) {
        float val = R[i] * cY[i];
        float rv  = rintf(val);
        float d   = val - rv;
        op[i * 8] = rv + d * d * d;
      }
    }

  // ---- chroma tiles: finish 2x2 mean while building B operand ---------
#pragma unroll
  for (int ch = 0; ch < 2; ++ch) {
    const float* src = ch ? sCr : sCb;
    v16h bX;
#pragma unroll
    for (int i = 0; i < 16; ++i) {
      float v = 0.25f * (src[(2 * i) * 17 + lc] + src[(2 * i + 1) * 17 + lc]);
      bX[i] = (_Float16)v;
    }
    v8f R = dct16(bX, aMt, bM, sT, lane);
    int by = wy * 2 + h;
    int bx = wx * 2 + bc;
    float* op = (ch ? crq : cbq) +
                (((size_t)b * 1024) + (size_t)by * 32 + bx) * 64 + vq;
#pragma unroll
    for (int i = 0; i < 8; ++i) {
      float val = R[i] * cC[i];
      float rv  = rintf(val);
      float d   = val - rv;
      op[i * 8] = rv + d * d * d;
    }
  }
}

extern "C" void kernel_launch(void* const* d_in, const int* in_sizes, int n_in,
                              void* d_out, int out_size, void* d_ws, size_t ws_size,
                              hipStream_t stream) {
  (void)in_sizes; (void)n_in; (void)out_size; (void)d_ws; (void)ws_size;
  const float* xin = (const float*)d_in[0];
  float* out = (float*)d_out;
  // 32 images * (512/32)^2 tiles = 8192 workgroups, one wave32 each.
  jpeg_enc_kernel<<<dim3(8192), dim3(32), 0, stream>>>(xin, out);
}
